// LanguageModel_3521873182874
// MI455X (gfx1250) — matrix operand order repaired and checked by reading, not compiled, not run on hardware
//
#include <hip/hip_runtime.h>
#include <hip/hip_bf16.h>
#include <math.h>

// ---- model dims ----
#define L_   12
#define H_   768
#define NH_  12
#define DH_  64
#define F_   3072
#define V_   21128
#define S_   512
#define B_   8
#define M_   (B_ * S_)   // 4096 rows
#define SEP_ 30
#define LN_EPS 1e-12f

typedef __attribute__((ext_vector_type(16))) _Float16 v16h;
typedef __attribute__((ext_vector_type(8)))  _Float16 v8h;
typedef __attribute__((ext_vector_type(8)))  float    v8f;

// LDS tile stride in halves: 40 (=80B, 16B-aligned, avoids bank conflicts)
#define LDS_STRIDE 40

// ---------------- gfx1250 async global->LDS copy (ASYNCcnt path) -------------
#if defined(__gfx1250__) && __has_builtin(__builtin_amdgcn_global_load_async_to_lds_b128)
#define HAS_ASYNC_LDS 1
#else
#define HAS_ASYNC_LDS 0
#endif

typedef int v4i_ __attribute__((vector_size(16)));
typedef __attribute__((address_space(1))) v4i_ GV4;
typedef __attribute__((address_space(3))) v4i_ LV4;

__device__ __forceinline__ void copy16_async(const void* g, void* l) {
#if HAS_ASYNC_LDS
  __builtin_amdgcn_global_load_async_to_lds_b128((GV4*)g, (LV4*)l, 0, 0);
#else
  *(uint4*)l = *(const uint4*)g;
#endif
}

__device__ __forceinline__ void wait_async() {
#if HAS_ASYNC_LDS
#if __has_builtin(__builtin_amdgcn_s_wait_asynccnt)
  __builtin_amdgcn_s_wait_asynccnt(0);
#else
  asm volatile("s_wait_asynccnt 0" ::: "memory");
#endif
#endif
}

// ---------------- WMMA fragment loaders (wave32 layouts, ISA 7.12.2) ---------
// A 16x32 f16: lane L holds row (L&15); half e -> k = (e/8)*16 + (L>>4)*8 + e%8
__device__ __forceinline__ v16h load_a_frag(const _Float16* tile, int r0, int lm, int lh) {
  const _Float16* base = tile + (r0 + lm) * LDS_STRIDE;
  v8h lo = *(const v8h*)(base + lh * 8);
  v8h hi = *(const v8h*)(base + 16 + lh * 8);
  v16h a;
#pragma unroll
  for (int i = 0; i < 8; ++i) { a[i] = lo[i]; a[i + 8] = hi[i]; }
  return a;
}
// B 32x16 f16 staged as [N][K] in LDS: lane L holds col (L&15); half e -> k = (e/8)*16 + (L>>4)*8 + e%8
__device__ __forceinline__ v16h load_b_frag(const _Float16* tile, int n0, int lm, int lh) {
  const _Float16* base = tile + (n0 + lm) * LDS_STRIDE;
  v8h lo = *(const v8h*)(base + lh * 8);
  v8h hi = *(const v8h*)(base + 16 + lh * 8);
  v16h b;
#pragma unroll
  for (int i = 0; i < 8; ++i) { b[i] = lo[i]; b[i + 8] = hi[i]; }
  return b;
}

// ---------------- Generic GEMM: out = A(f16)[M,K] x W(fp32)[K,N] + bias ------
// 256 threads = 8 waves; block tile 128x64, K-step 32; wave tile 32x32 (2x2 WMMA)
// EPI: 0 = none, 1 = exact GELU
template <int EPI>
__global__ __launch_bounds__(256) void gemm_wf16(
    const _Float16* __restrict__ A, const float* __restrict__ W,
    const float* __restrict__ bias,
    float* __restrict__ out32, _Float16* __restrict__ out16,
    int M, int N, int K, int lda, int ldb, int ldc) {
  __shared__ _Float16 As[128 * LDS_STRIDE];
  __shared__ _Float16 Bs[64 * LDS_STRIDE];
  const int t  = threadIdx.x;
  const int mb = blockIdx.y * 128;
  const int nb = blockIdx.x * 64;
  const int w = t >> 5, lane = t & 31;
  const int wm = w & 3, wn = w >> 2;           // 4x2 wave grid
  const int lm = lane & 15, lh = lane >> 4;
  const int wr = wm * 32, wc = wn * 32;

  v8f acc[2][2];
#pragma unroll
  for (int mi = 0; mi < 2; ++mi)
#pragma unroll
    for (int ni = 0; ni < 2; ++ni) acc[mi][ni] = (v8f)0.0f;

  for (int kb = 0; kb < K; kb += 32) {
    // stage A: 128x32 halves, async direct global->LDS (b128 per thread)
#pragma unroll
    for (int i = 0; i < 2; ++i) {
      int c = t + i * 256;
      int row = c >> 2, ch = c & 3;
      copy16_async(A + (size_t)(mb + row) * lda + kb + ch * 8,
                   &As[row * LDS_STRIDE + ch * 8]);
    }
    // stage B: fp32 weights [K][N] -> f16 LDS [N][K] (transposed), N-guarded
#pragma unroll
    for (int j = 0; j < 8; ++j) {
      int e = t + j * 256;
      int k = e >> 6, n = e & 63;
      float v = 0.0f;
      if (nb + n < N) v = W[(size_t)(kb + k) * ldb + nb + n];
      Bs[n * LDS_STRIDE + k] = (_Float16)v;
      // prefetch next K-step's weight row into GL2 to hide fp32 latency
      if (kb + 32 < K && nb + n < N)
        __builtin_prefetch(W + (size_t)(kb + 32 + k) * ldb + nb + n, 0, 1);
    }
    wait_async();
    __syncthreads();
    v16h af[2], bf[2];
#pragma unroll
    for (int mi = 0; mi < 2; ++mi) af[mi] = load_a_frag(As, wr + mi * 16, lm, lh);
#pragma unroll
    for (int ni = 0; ni < 2; ++ni) bf[ni] = load_b_frag(Bs, wc + ni * 16, lm, lh);
#pragma unroll
    for (int mi = 0; mi < 2; ++mi)
#pragma unroll
      for (int ni = 0; ni < 2; ++ni)
        acc[mi][ni] = __builtin_amdgcn_wmma_f32_16x16x32_f16(
            false, af[mi], false, bf[ni], (short)0, acc[mi][ni], false, false);
    __syncthreads();
  }

#pragma unroll
  for (int mi = 0; mi < 2; ++mi)
#pragma unroll
    for (int ni = 0; ni < 2; ++ni) {
      int col = nb + wc + ni * 16 + lm;
      if (col >= N) continue;
      float bv = bias ? bias[col] : 0.0f;
#pragma unroll
      for (int r = 0; r < 8; ++r) {
        int row = mb + wr + mi * 16 + lh * 8 + r;
        float v = acc[mi][ni][r] + bv;
        if (EPI == 1) v = 0.5f * v * (1.0f + erff(v * 0.70710678118f));
        if (out32) out32[(size_t)row * ldc + col] = v;
        if (out16) out16[(size_t)row * ldc + col] = (_Float16)v;
      }
    }
}

// ---------------- Attention scores: S = (Q Kt)*scale + mask, per (b,h) ------
__global__ __launch_bounds__(256) void attn_scores_k(
    const _Float16* __restrict__ q16, const _Float16* __restrict__ k16,
    const int* __restrict__ x, _Float16* __restrict__ scores) {
  __shared__ _Float16 As[128 * LDS_STRIDE];
  __shared__ _Float16 Bs[64 * LDS_STRIDE];
  const int bh = blockIdx.z;
  const int b = bh / NH_, h = bh % NH_;
  const _Float16* Aq = q16 + (size_t)b * S_ * H_ + h * DH_;
  const _Float16* Bk = k16 + (size_t)b * S_ * H_ + h * DH_;
  const int t  = threadIdx.x;
  const int mb = blockIdx.y * 128;
  const int nb = blockIdx.x * 64;
  const int w = t >> 5, lane = t & 31;
  const int wm = w & 3, wn = w >> 2;
  const int lm = lane & 15, lh = lane >> 4;
  const int wr = wm * 32, wc = wn * 32;

  v8f acc[2][2];
#pragma unroll
  for (int mi = 0; mi < 2; ++mi)
#pragma unroll
    for (int ni = 0; ni < 2; ++ni) acc[mi][ni] = (v8f)0.0f;

#pragma unroll
  for (int kb = 0; kb < DH_; kb += 32) {
#pragma unroll
    for (int i = 0; i < 2; ++i) {
      int c = t + i * 256;
      int row = c >> 2, ch = c & 3;
      copy16_async(Aq + (size_t)(mb + row) * H_ + kb + ch * 8,
                   &As[row * LDS_STRIDE + ch * 8]);
    }
    { // Kt stage: reading K row-major == staging B transposed [N][K]
      int n = t >> 2, ch = t & 3;
      copy16_async(Bk + (size_t)(nb + n) * H_ + kb + ch * 8,
                   &Bs[n * LDS_STRIDE + ch * 8]);
    }
    wait_async();
    __syncthreads();
    v16h af[2], bf[2];
#pragma unroll
    for (int mi = 0; mi < 2; ++mi) af[mi] = load_a_frag(As, wr + mi * 16, lm, lh);
#pragma unroll
    for (int ni = 0; ni < 2; ++ni) bf[ni] = load_b_frag(Bs, wc + ni * 16, lm, lh);
#pragma unroll
    for (int mi = 0; mi < 2; ++mi)
#pragma unroll
      for (int ni = 0; ni < 2; ++ni)
        acc[mi][ni] = __builtin_amdgcn_wmma_f32_16x16x32_f16(
            false, af[mi], false, bf[ni], (short)0, acc[mi][ni], false, false);
    __syncthreads();
  }

#pragma unroll
  for (int mi = 0; mi < 2; ++mi)
#pragma unroll
    for (int ni = 0; ni < 2; ++ni) {
      int col = nb + wc + ni * 16 + lm;
      bool colpad = (x[b * S_ + col] != 0);
#pragma unroll
      for (int r = 0; r < 8; ++r) {
        int row = mb + wr + mi * 16 + lh * 8 + r;
        bool ok = colpad && (x[b * S_ + row] != 0) &&
                  ((col <= row) || ((row <= SEP_) && (col <= SEP_)));
        float v = ok ? acc[mi][ni][r] * 0.125f : -30000.0f;
        scores[((size_t)bh * S_ + row) * S_ + col] = (_Float16)v;
      }
    }
}

// ---------------- ctx = P[S,S] x V[S,DH], per (b,h) --------------------------
__global__ __launch_bounds__(256) void attn_ctx_k(
    const _Float16* __restrict__ probs, const _Float16* __restrict__ vmat,
    _Float16* __restrict__ ctx16) {
  __shared__ _Float16 As[128 * LDS_STRIDE];
  __shared__ _Float16 Bs[64 * LDS_STRIDE];
  const int bh = blockIdx.z;
  const int b = bh / NH_, h = bh % NH_;
  const _Float16* Ap = probs + (size_t)bh * S_ * S_;
  const _Float16* Bv = vmat + (size_t)b * S_ * H_ + h * DH_;
  const int t  = threadIdx.x;
  const int mb = blockIdx.y * 128;
  const int w = t >> 5, lane = t & 31;
  const int wm = w & 3, wn = w >> 2;
  const int lm = lane & 15, lh = lane >> 4;
  const int wr = wm * 32, wc = wn * 32;

  v8f acc[2][2];
#pragma unroll
  for (int mi = 0; mi < 2; ++mi)
#pragma unroll
    for (int ni = 0; ni < 2; ++ni) acc[mi][ni] = (v8f)0.0f;

  for (int kb = 0; kb < S_; kb += 32) {
#pragma unroll
    for (int i = 0; i < 2; ++i) {
      int c = t + i * 256;
      int row = c >> 2, ch = c & 3;
      copy16_async(Ap + (size_t)(mb + row) * S_ + kb + ch * 8,
                   &As[row * LDS_STRIDE + ch * 8]);
    }
    { // V is [K][N] f16, transpose into LDS [N][K]
      int k = t >> 3, n0 = (t & 7) * 8;
      v8h u = *(const v8h*)(Bv + (size_t)(kb + k) * H_ + n0);
#pragma unroll
      for (int j = 0; j < 8; ++j) Bs[(n0 + j) * LDS_STRIDE + k] = u[j];
    }
    wait_async();
    __syncthreads();
    v16h af[2], bf[2];
#pragma unroll
    for (int mi = 0; mi < 2; ++mi) af[mi] = load_a_frag(As, wr + mi * 16, lm, lh);
#pragma unroll
    for (int ni = 0; ni < 2; ++ni) bf[ni] = load_b_frag(Bs, wc + ni * 16, lm, lh);
#pragma unroll
    for (int mi = 0; mi < 2; ++mi)
#pragma unroll
      for (int ni = 0; ni < 2; ++ni)
        acc[mi][ni] = __builtin_amdgcn_wmma_f32_16x16x32_f16(
            false, af[mi], false, bf[ni], (short)0, acc[mi][ni], false, false);
    __syncthreads();
  }

#pragma unroll
  for (int mi = 0; mi < 2; ++mi)
#pragma unroll
    for (int ni = 0; ni < 2; ++ni) {
      int col = wc + ni * 16 + lm;       // < 64
#pragma unroll
      for (int r = 0; r < 8; ++r) {
        int row = mb + wr + mi * 16 + lh * 8 + r;
        ctx16[((size_t)b * S_ + row) * H_ + h * DH_ + col] = (_Float16)acc[mi][ni][r];
      }
    }
}

// ---------------- embeddings + LayerNorm -------------------------------------
__global__ __launch_bounds__(256) void embed_ln_k(
    const int* __restrict__ x, const float* __restrict__ we,
    const float* __restrict__ pe, const float* __restrict__ te,
    const float* __restrict__ g, const float* __restrict__ bb,
    float* __restrict__ h32, _Float16* __restrict__ h16) {
  __shared__ float sS[256], sQ[256];
  const int row = blockIdx.x, s = row % S_, t = threadIdx.x;
  const int tok = x[row];
  float vals[3], sum = 0.f, sq = 0.f;
#pragma unroll
  for (int i = 0; i < 3; ++i) {
    int j = t + i * 256;
    float v = we[(size_t)tok * H_ + j] + pe[(size_t)s * H_ + j] + te[j];
    vals[i] = v; sum += v; sq += v * v;
  }
  sS[t] = sum; sQ[t] = sq; __syncthreads();
  for (int o = 128; o > 0; o >>= 1) {
    if (t < o) { sS[t] += sS[t + o]; sQ[t] += sQ[t + o]; }
    __syncthreads();
  }
  float mean = sS[0] / (float)H_;
  float var  = sQ[0] / (float)H_ - mean * mean;
  float inv  = rsqrtf(var + LN_EPS);
#pragma unroll
  for (int i = 0; i < 3; ++i) {
    int j = t + i * 256;
    float o = (vals[i] - mean) * inv * g[j] + bb[j];
    h32[(size_t)row * H_ + j] = o;
    h16[(size_t)row * H_ + j] = (_Float16)o;
  }
}

// ---------------- h = LN(h + t) ----------------------------------------------
__global__ __launch_bounds__(256) void add_ln_k(
    float* __restrict__ h32, const float* __restrict__ t32,
    const float* __restrict__ g, const float* __restrict__ bb,
    _Float16* __restrict__ h16) {
  __shared__ float sS[256], sQ[256];
  const int row = blockIdx.x, t = threadIdx.x;
  float vals[3], sum = 0.f, sq = 0.f;
#pragma unroll
  for (int i = 0; i < 3; ++i) {
    int j = t + i * 256;
    float v = h32[(size_t)row * H_ + j] + t32[(size_t)row * H_ + j];
    vals[i] = v; sum += v; sq += v * v;
  }
  sS[t] = sum; sQ[t] = sq; __syncthreads();
  for (int o = 128; o > 0; o >>= 1) {
    if (t < o) { sS[t] += sS[t + o]; sQ[t] += sQ[t + o]; }
    __syncthreads();
  }
  float mean = sS[0] / (float)H_;
  float var  = sQ[0] / (float)H_ - mean * mean;
  float inv  = rsqrtf(var + LN_EPS);
#pragma unroll
  for (int i = 0; i < 3; ++i) {
    int j = t + i * 256;
    float o = (vals[i] - mean) * inv * g[j] + bb[j];
    h32[(size_t)row * H_ + j] = o;
    h16[(size_t)row * H_ + j] = (_Float16)o;
  }
}

// ---------------- softmax over 512-wide f16 rows (in place) ------------------
__global__ __launch_bounds__(256) void softmax512_k(_Float16* __restrict__ sc) {
  __shared__ float red[256];
  const int t = threadIdx.x;
  _Float16* row = sc + (size_t)blockIdx.x * 512;
  float v0 = (float)row[t], v1 = (float)row[t + 256];
  red[t] = fmaxf(v0, v1); __syncthreads();
  for (int o = 128; o > 0; o >>= 1) {
    if (t < o) red[t] = fmaxf(red[t], red[t + o]);
    __syncthreads();
  }
  float m = red[0]; __syncthreads();
  float e0 = __expf(v0 - m), e1 = __expf(v1 - m);
  red[t] = e0 + e1; __syncthreads();
  for (int o = 128; o > 0; o >>= 1) {
    if (t < o) red[t] += red[t + o];
    __syncthreads();
  }
  float inv = 1.0f / red[0];
  row[t]       = (_Float16)(e0 * inv);
  row[t + 256] = (_Float16)(e1 * inv);
}

// ---------------- softmax over V-wide fp32 rows (in place in d_out) ----------
__global__ __launch_bounds__(256) void softmaxV_k(float* __restrict__ out) {
  __shared__ float red[256];
  const int t = threadIdx.x;
  float* row = out + (size_t)blockIdx.x * V_;
  float m = -3.4e38f;
  for (int j = t; j < V_; j += 256) m = fmaxf(m, row[j]);
  red[t] = m; __syncthreads();
  for (int o = 128; o > 0; o >>= 1) {
    if (t < o) red[t] = fmaxf(red[t], red[t + o]);
    __syncthreads();
  }
  m = red[0]; __syncthreads();
  float s = 0.f;
  for (int j = t; j < V_; j += 256) s += __expf(row[j] - m);
  red[t] = s; __syncthreads();
  for (int o = 128; o > 0; o >>= 1) {
    if (t < o) red[t] += red[t + o];
    __syncthreads();
  }
  float inv = 1.0f / red[0];
  for (int j = t; j < V_; j += 256) row[j] = __expf(row[j] - m) * inv;
}

// =============================================================================
extern "C" void kernel_launch(void* const* d_in, const int* in_sizes, int n_in,
                              void* d_out, int out_size, void* d_ws, size_t ws_size,
                              hipStream_t stream) {
  const int*   x     = (const int*)d_in[0];
  const float* we    = (const float*)d_in[1];
  const float* pe    = (const float*)d_in[2];
  const float* te    = (const float*)d_in[3];
  const float* lneg  = (const float*)d_in[4];
  const float* lneb  = (const float*)d_in[5];
  const float* WqA   = (const float*)d_in[6];
  const float* bqA   = (const float*)d_in[7];
  const float* WkA   = (const float*)d_in[8];
  const float* bkA   = (const float*)d_in[9];
  const float* WvA   = (const float*)d_in[10];
  const float* bvA   = (const float*)d_in[11];
  const float* WoA   = (const float*)d_in[12];
  const float* boA   = (const float*)d_in[13];
  const float* ln1g  = (const float*)d_in[14];
  const float* ln1b  = (const float*)d_in[15];
  const float* W1A   = (const float*)d_in[16];
  const float* b1A   = (const float*)d_in[17];
  const float* W2A   = (const float*)d_in[18];
  const float* b2A   = (const float*)d_in[19];
  const float* ln2g  = (const float*)d_in[20];
  const float* ln2b  = (const float*)d_in[21];
  const float* clsW  = (const float*)d_in[22];
  const float* clsb  = (const float*)d_in[23];
  float* out = (float*)d_out;

  char* ws = (char*)d_ws;
  size_t off = 0;
  auto alloc = [&](size_t bytes) -> char* {
    char* p = ws + off;
    off += (bytes + 255) & ~(size_t)255;
    return p;
  };
  float*    h32   = (float*)   alloc((size_t)M_ * H_ * 4);
  _Float16* h16   = (_Float16*)alloc((size_t)M_ * H_ * 2);
  _Float16* q16   = (_Float16*)alloc((size_t)M_ * H_ * 2);
  _Float16* k16   = (_Float16*)alloc((size_t)M_ * H_ * 2);
  _Float16* v16p  = (_Float16*)alloc((size_t)M_ * H_ * 2);
  _Float16* ctx16 = (_Float16*)alloc((size_t)M_ * H_ * 2);
  float*    t32   = (float*)   alloc((size_t)M_ * H_ * 4);
  _Float16* f116  = (_Float16*)alloc((size_t)M_ * F_ * 2);
  _Float16* sc16  = (_Float16*)alloc((size_t)B_ * NH_ * S_ * S_ * 2);
  (void)ws_size; (void)in_sizes; (void)n_in; (void)out_size;

  // embeddings + LN
  embed_ln_k<<<M_, 256, 0, stream>>>(x, we, pe, te, lneg, lneb, h32, h16);

  const dim3 blk(256);
  const dim3 gHH((H_ + 63) / 64, M_ / 128);
  const dim3 gHF((F_ + 63) / 64, M_ / 128);
  const dim3 gFH((H_ + 63) / 64, M_ / 128);
  const dim3 gSc(S_ / 64, S_ / 128, B_ * NH_);
  const dim3 gCx(1, S_ / 128, B_ * NH_);

  for (int l = 0; l < L_; ++l) {
    const float* Wq = WqA + (size_t)l * H_ * H_;
    const float* Wk = WkA + (size_t)l * H_ * H_;
    const float* Wv = WvA + (size_t)l * H_ * H_;
    const float* Wo = WoA + (size_t)l * H_ * H_;
    const float* W1 = W1A + (size_t)l * H_ * F_;
    const float* W2 = W2A + (size_t)l * F_ * H_;
    const float* bq = bqA + (size_t)l * H_;
    const float* bk = bkA + (size_t)l * H_;
    const float* bv = bvA + (size_t)l * H_;
    const float* bo = boA + (size_t)l * H_;
    const float* b1 = b1A + (size_t)l * F_;
    const float* b2 = b2A + (size_t)l * H_;
    const float* g1 = ln1g + (size_t)l * H_;
    const float* gb1 = ln1b + (size_t)l * H_;
    const float* g2 = ln2g + (size_t)l * H_;
    const float* gb2 = ln2b + (size_t)l * H_;

    // QKV projections (f16 outputs, bias fused)
    gemm_wf16<0><<<gHH, blk, 0, stream>>>(h16, Wq, bq, nullptr, q16, M_, H_, H_, H_, H_, H_);
    gemm_wf16<0><<<gHH, blk, 0, stream>>>(h16, Wk, bk, nullptr, k16, M_, H_, H_, H_, H_, H_);
    gemm_wf16<0><<<gHH, blk, 0, stream>>>(h16, Wv, bv, nullptr, v16p, M_, H_, H_, H_, H_, H_);

    // attention: scores (+mask) -> softmax -> ctx
    attn_scores_k<<<gSc, blk, 0, stream>>>(q16, k16, x, sc16);
    softmax512_k<<<B_ * NH_ * S_, blk, 0, stream>>>(sc16);
    attn_ctx_k<<<gCx, blk, 0, stream>>>(sc16, v16p, ctx16);

    // output proj + residual LN
    gemm_wf16<0><<<gHH, blk, 0, stream>>>(ctx16, Wo, bo, t32, nullptr, M_, H_, H_, H_, H_, H_);
    add_ln_k<<<M_, blk, 0, stream>>>(h32, t32, g1, gb1, h16);

    // FFN: GELU(h W1 + b1) W2 + b2, residual LN
    gemm_wf16<1><<<gHF, blk, 0, stream>>>(h16, W1, b1, nullptr, f116, M_, F_, H_, H_, F_, F_);
    gemm_wf16<0><<<gFH, blk, 0, stream>>>(f116, W2, b2, t32, nullptr, M_, H_, F_, F_, H_, H_);
    add_ln_k<<<M_, blk, 0, stream>>>(h32, t32, g2, gb2, h16);
  }

  // classifier: logits straight into d_out, then in-place softmax over V
  const dim3 gV((V_ + 63) / 64, M_ / 128);
  gemm_wf16<0><<<gV, blk, 0, stream>>>(h16, clsW, clsb, out, nullptr, M_, V_, H_, H_, V_, V_);
  softmaxV_k<<<M_, blk, 0, stream>>>(out);
}
